// Attention_11519102288633
// MI455X (gfx1250) — matrix operand order repaired and checked by reading, not compiled, not run on hardware
//
#include <hip/hip_runtime.h>
#include <hip/hip_bf16.h>

// Problem constants (match reference)
#define B_  2
#define T_  2048
#define D_  512
#define H_  8
#define DH_ 64
#define W_  64

typedef __attribute__((ext_vector_type(16))) __bf16 v16bf;
typedef __attribute__((ext_vector_type(8)))  __bf16 v8bf;
typedef __attribute__((ext_vector_type(8)))  float  v8f;
typedef __attribute__((ext_vector_type(4)))  unsigned int v4ui;
typedef __attribute__((ext_vector_type(8)))  unsigned int v8ui;

__device__ inline v16bf zero16bf() {
  v16bf z;
#pragma unroll
  for (int i = 0; i < 16; ++i) z[i] = (__bf16)0.0f;
  return z;
}

__device__ inline v8f wmma_bf16(v16bf a, v16bf b, v8f c) {
  // (neg_a, A, neg_b, B, c_mod, C, reuse_a, reuse_b)
  return __builtin_amdgcn_wmma_f32_16x16x32_bf16(false, a, false, b,
                                                 (short)0, c, false, false);
}

// ---- Tensor Data Mover: 2-D tile load (global -> LDS), 2-group D#.
// data_size = 4B; tile = row_dwords x rows; row stride = stride_dwords.
__device__ inline void tdm_load_2d(const void* gaddr, void* lds_ptr,
                                   unsigned rows, unsigned row_dwords,
                                   unsigned stride_dwords) {
  unsigned long long ga = (unsigned long long)(uintptr_t)gaddr;
  unsigned lo32 = (unsigned)(uintptr_t)lds_ptr;  // low 32 bits = LDS byte addr
  v4ui g0;
  g0[0] = 1u;                                   // count=1, user descriptor
  g0[1] = lo32;                                 // lds_addr
  g0[2] = (unsigned)ga;                         // global_addr[31:0]
  g0[3] = (unsigned)((ga >> 32) & 0x1FFFFFFu)   // global_addr[56:32]
          | (2u << 30);                         // type = 2 ("image")
  v8ui g1;
  g1[0] = 2u << 16;                             // data_size = 4 bytes
  g1[1] = row_dwords << 16;                     // tensor_dim0[15:0] @ bit48
  g1[2] = rows << 16;                           // tensor_dim1[15:0] @ bit80
  g1[3] = row_dwords << 16;                     // tile_dim0 @ bit112
  g1[4] = rows & 0xFFFFu;                       // tile_dim1 @ bit128
  g1[5] = stride_dwords;                        // tensor_dim0_stride[31:0]
  g1[6] = 0u;
  g1[7] = 0u;
  asm volatile("tensor_load_to_lds %0, %1" ::"s"(g0), "s"(g1) : "memory");
}

// ---------------------------------------------------------------- prep kernels
__global__ void cvt_bf16_kernel(const float* __restrict__ in,
                                __bf16* __restrict__ out, int n) {
  int i = blockIdx.x * blockDim.x + threadIdx.x;
  if (i < n) out[i] = (__bf16)in[i];
}

// out[n*K + k] = (bf16) in[k*N + n]   (store B^T so B-columns are contiguous)
__global__ void transpose_cvt_kernel(const float* __restrict__ in,
                                     __bf16* __restrict__ out, int N, int K) {
  int i = blockIdx.x * blockDim.x + threadIdx.x;
  if (i >= N * K) return;
  int n = i / K;
  int k = i - n * K;
  out[(size_t)n * K + k] = (__bf16)in[(size_t)k * N + n];
}

// ---------------------------------------------------------------- WMMA GEMM
// C[M,N] = A[M,K] (bf16, row-major) * Bt[N,K] (bf16, pre-transposed) (+ bias)
// Block = 4 waves = 64x64 output tile; each wave computes a 16x64 strip, so
// one A fragment feeds 4 WMMAs per K-step.
__global__ void wmma_gemm_kernel(const __bf16* __restrict__ A,
                                 const __bf16* __restrict__ Bt,
                                 const float* __restrict__ bias,
                                 float* __restrict__ C,
                                 int M, int N, int K) {
  int lane = threadIdx.x & 31;
  int wave = threadIdx.x >> 5;
  int lo = lane & 15, hi = lane >> 4;
  int ntn = N >> 6;  // 64-wide column tiles
  int tm  = (blockIdx.x / ntn) * 64 + wave * 16;
  int tn0 = (blockIdx.x % ntn) * 64;
  if (tm >= M) return;

  const __bf16* ap = A + (size_t)(tm + lo) * K;  // A row for this lane
  v8f acc[4];
#pragma unroll
  for (int nt = 0; nt < 4; ++nt) acc[nt] = (v8f){};

  for (int kk = 0; kk < K; kk += 32) {
    // A fragment: lanes 0-15 hold K kk..kk+15; lanes 16-31 hold kk+16..kk+31
    v8bf a0 = *(const v8bf*)(ap + kk + (hi << 4));
    v8bf a1 = *(const v8bf*)(ap + kk + 8 + (hi << 4));
    v16bf av;
#pragma unroll
    for (int e = 0; e < 8; ++e) { av[e] = a0[e]; av[e + 8] = a1[e]; }
#pragma unroll
    for (int nt = 0; nt < 4; ++nt) {
      // B fragment: lane's column contiguous K chunk from pre-transposed Bt
      v16bf bv = *(const v16bf*)(Bt + (size_t)(tn0 + nt * 16 + lo) * K + kk +
                                 (hi << 4));
      acc[nt] = wmma_bf16(av, bv, acc[nt]);
    }
  }
  int orow = tm + (hi << 3);
#pragma unroll
  for (int nt = 0; nt < 4; ++nt) {
    int col = tn0 + nt * 16 + lo;
    float badd = bias ? bias[col] : 0.0f;
#pragma unroll
    for (int r = 0; r < 8; ++r)
      C[(size_t)(orow + r) * N + col] = acc[nt][r] + badd;
  }
}

// ---------------------------------------------------------------- RoPE+permute
// qkv_raw: (B*T, 1536) f32  [q | k | v]
// qb/kb: (B,H,T,64) bf16 ; vT: (B,H,64,T) bf16
__global__ void rope_permute_kernel(const float* __restrict__ qkv,
                                    __bf16* __restrict__ qb,
                                    __bf16* __restrict__ kb,
                                    __bf16* __restrict__ vT) {
  int i = blockIdx.x * blockDim.x + threadIdx.x;
  if (i >= B_ * T_ * (D_ / 2)) return;
  int d = i & 255;
  int t = (i >> 8) & (T_ - 1);
  int b = i >> 19;

  const float* row = qkv + (size_t)(b * T_ + t) * 1536;
  // inv_freq = 10000^(-d/256) = exp(-d * ln(1e4)/256)
  float inv_freq = __expf(-(float)d * (9.210340371976184f / 256.0f));
  float ang = (float)t * inv_freq;
  float c = cosf(ang), s = sinf(ang);

  float q1 = row[d],        q2 = row[d + 256];
  float k1 = row[512 + d],  k2 = row[768 + d];
  float v1 = row[1024 + d], v2 = row[1280 + d];

  int h1 = d >> 6, dd = d & 63;
  int h2 = h1 + 4;  // (d+256)>>6
  size_t bh1 = (size_t)(b * H_ + h1), bh2 = (size_t)(b * H_ + h2);

  qb[(bh1 * T_ + t) * DH_ + dd] = (__bf16)(q1 * c - q2 * s);
  qb[(bh2 * T_ + t) * DH_ + dd] = (__bf16)(q1 * s + q2 * c);
  kb[(bh1 * T_ + t) * DH_ + dd] = (__bf16)(k1 * c - k2 * s);
  kb[(bh2 * T_ + t) * DH_ + dd] = (__bf16)(k1 * s + k2 * c);
  vT[(bh1 * DH_ + dd) * T_ + t] = (__bf16)v1;
  vT[(bh2 * DH_ + dd) * T_ + t] = (__bf16)v2;
}

// ---------------------------------------------------------------- attention
// One wave per (b*H+h, 16-query block). Window keys j in [t-63, t]; keys j<0
// are zero vectors (score 0, in the softmax) per the reference's zero-padding.
// Q tile and the 96-row K window strip are staged into LDS by the TDM.
__global__ void attn_kernel(const __bf16* __restrict__ qb,
                            const __bf16* __restrict__ kb,
                            const __bf16* __restrict__ vT,
                            __bf16* __restrict__ ao) {
  __shared__ __align__(16) __bf16 shQ[16 * 64];  // 16 query rows
  __shared__ __align__(16) __bf16 shK[96 * 64];  // key rows j = t0-64+jj
  __shared__ __align__(16) __bf16 shA[16 * 96];  // attn re-stripe buffer

  int lane = threadIdx.x & 31;
  int lo = lane & 15, hi = lane >> 4;
  int t0 = blockIdx.x << 4;
  int bh = blockIdx.y;  // b*H + h
  int b  = bh >> 3;
  int h  = bh & 7;

  // ---- zero-fill out-of-range key rows, then TDM the in-range strip
  int j0   = t0 - 64 < 0 ? 0 : t0 - 64;        // first valid key row
  int j1   = t0 + 32 > T_ ? T_ : t0 + 32;      // one past last valid key row
  int head = j0 - (t0 - 64);                   // leading zero rows
  int tail = (t0 + 32) - j1;                   // trailing zero rows
  uint4 z4 = make_uint4(0u, 0u, 0u, 0u);
  for (int c = lane; c < head * 8; c += 32) ((uint4*)shK)[c] = z4;
  for (int c = lane; c < tail * 8; c += 32)
    ((uint4*)shK)[(96 - tail) * 8 + c] = z4;

  tdm_load_2d(qb + ((size_t)bh * T_ + t0) * DH_, shQ, 16u, 32u, 32u);
  tdm_load_2d(kb + ((size_t)bh * T_ + j0) * DH_, shK + (size_t)head * DH_,
              (unsigned)(j1 - j0), 32u, 32u);
  __builtin_amdgcn_s_wait_tensorcnt(0);
  asm volatile("" ::: "memory");

  // ---- Q fragments (rows t0..t0+15, K = head dim 0..63, two K=32 steps)
  const __bf16* qp = shQ + lo * DH_;
  v16bf aq[2];
#pragma unroll
  for (int ks = 0; ks < 2; ++ks) {
    v8bf a0 = *(const v8bf*)(qp + ks * 32 + (hi << 4));
    v8bf a1 = *(const v8bf*)(qp + ks * 32 + 8 + (hi << 4));
#pragma unroll
    for (int e = 0; e < 8; ++e) { aq[ks][e] = a0[e]; aq[ks][e + 8] = a1[e]; }
  }

  // ---- scores: 16x96 strip (6 tiles x K=64); OOB keys are zero rows in LDS
  v8f sc[6];
#pragma unroll
  for (int jt = 0; jt < 6; ++jt) {
    const __bf16* kp = shK + (jt * 16 + lo) * DH_;  // one key row per lane
    v16bf b0 = *(const v16bf*)(kp + (hi << 4));
    v16bf b1 = *(const v16bf*)(kp + 32 + (hi << 4));
    v8f s = {};
    s = wmma_bf16(aq[0], b0, s);
    s = wmma_bf16(aq[1], b1, s);
    sc[jt] = s;
  }

  // ---- masked softmax over the 64-wide window, f32, per row
  const float scale = 0.125f;  // 1/sqrt(64)
#pragma unroll
  for (int r = 0; r < 8; ++r) {
    int t = t0 + (hi << 3) + r;  // this lane-half's query row
    float m = -3.0e38f;
#pragma unroll
    for (int jt = 0; jt < 6; ++jt) {
      int j = t0 - 64 + jt * 16 + lo;
      bool valid = (j <= t) && (j >= t - (W_ - 1));
      float x = valid ? sc[jt][r] * scale : -3.0e38f;
      sc[jt][r] = x;
      m = fmaxf(m, x);
    }
#pragma unroll
    for (int mk = 1; mk < 16; mk <<= 1) m = fmaxf(m, __shfl_xor(m, mk, 32));
    float sum = 0.0f;
#pragma unroll
    for (int jt = 0; jt < 6; ++jt) {
      float p = (sc[jt][r] > -1.0e38f) ? __expf(sc[jt][r] - m) : 0.0f;
      sc[jt][r] = p;
      sum += p;
    }
#pragma unroll
    for (int mk = 1; mk < 16; mk <<= 1) sum += __shfl_xor(sum, mk, 32);
    float rinv = 1.0f / sum;
#pragma unroll
    for (int jt = 0; jt < 6; ++jt) sc[jt][r] *= rinv;
  }

  // ---- re-stripe attn (D-layout) -> A-layout via LDS, as bf16
#pragma unroll
  for (int jt = 0; jt < 6; ++jt)
#pragma unroll
    for (int r = 0; r < 8; ++r)
      shA[((hi << 3) + r) * 96 + jt * 16 + lo] = (__bf16)sc[jt][r];
  __syncthreads();

  // ---- out = attn(16x96) @ Vwin(96x64), three K=32 steps, 4 N-tiles
  v8f acco[4];
#pragma unroll
  for (int nt = 0; nt < 4; ++nt) acco[nt] = (v8f){};
#pragma unroll
  for (int ks = 0; ks < 3; ++ks) {
    const __bf16* sp = shA + lo * 96 + ks * 32 + (hi << 4);
    v8bf a0 = *(const v8bf*)sp;
    v8bf a1 = *(const v8bf*)(sp + 8);
    v16bf av;
#pragma unroll
    for (int e = 0; e < 8; ++e) { av[e] = a0[e]; av[e + 8] = a1[e]; }
    int jstart = t0 - 64 + ks * 32 + (hi << 4);  // 16 contiguous keys per lane
#pragma unroll
    for (int nt = 0; nt < 4; ++nt) {
      v16bf bv = zero16bf();
      if (jstart >= 0 && jstart + 16 <= T_) {
        const __bf16* vp = vT + ((size_t)bh * DH_ + nt * 16 + lo) * T_ + jstart;
        bv = *(const v16bf*)vp;
      }
      acco[nt] = wmma_bf16(av, bv, acco[nt]);
    }
  }

  // ---- store to (B,T,512) bf16 for the output projection
#pragma unroll
  for (int nt = 0; nt < 4; ++nt)
#pragma unroll
    for (int r = 0; r < 8; ++r) {
      int t = t0 + (hi << 3) + r;
      ao[(size_t)(b * T_ + t) * D_ + h * DH_ + nt * 16 + lo] =
          (__bf16)acco[nt][r];
    }
}

// ---------------------------------------------------------------- launcher
extern "C" void kernel_launch(void* const* d_in, const int* in_sizes, int n_in,
                              void* d_out, int out_size, void* d_ws,
                              size_t ws_size, hipStream_t stream) {
  const float* x    = (const float*)d_in[0];
  const float* Wq   = (const float*)d_in[1];
  const float* Wkv  = (const float*)d_in[2];
  const float* Wlin = (const float*)d_in[3];
  const float* blin = (const float*)d_in[4];
  float* out = (float*)d_out;

  char*  ws  = (char*)d_ws;
  size_t off = 0;
  auto alloc = [&](size_t bytes) -> void* {
    void* p = ws + off;
    off = (off + bytes + 255) & ~(size_t)255;
    return p;
  };

  const int MT = B_ * T_;  // 4096 rows
  __bf16* xb     = (__bf16*)alloc((size_t)MT * D_ * 2);
  __bf16* WqkvT  = (__bf16*)alloc((size_t)(3 * D_) * D_ * 2);  // 1536 x 512
  __bf16* WlinT  = (__bf16*)alloc((size_t)D_ * D_ * 2);
  float*  qkvraw = (float*)alloc((size_t)MT * (3 * D_) * 4);
  __bf16* qbuf   = (__bf16*)alloc((size_t)B_ * H_ * T_ * DH_ * 2);
  __bf16* kbuf   = (__bf16*)alloc((size_t)B_ * H_ * T_ * DH_ * 2);
  __bf16* vTbuf  = (__bf16*)alloc((size_t)B_ * H_ * DH_ * T_ * 2);
  __bf16* ao     = (__bf16*)alloc((size_t)MT * D_ * 2);

  // 1) convert x -> bf16
  {
    int n = MT * D_;
    cvt_bf16_kernel<<<(n + 255) / 256, 256, 0, stream>>>(x, xb, n);
  }
  // 2) transpose+convert weights (Wq and Wkv packed contiguously as B^T rows)
  {
    int n = D_ * D_;
    transpose_cvt_kernel<<<(n + 255) / 256, 256, 0, stream>>>(Wq, WqkvT, D_, D_);
    int n2 = 2 * D_ * D_;
    transpose_cvt_kernel<<<(n2 + 255) / 256, 256, 0, stream>>>(
        Wkv, WqkvT + (size_t)D_ * D_, 2 * D_, D_);
    transpose_cvt_kernel<<<(n + 255) / 256, 256, 0, stream>>>(Wlin, WlinT, D_,
                                                              D_);
  }
  // 3) fused QKV projection: (4096x512) @ (512x1536) -> f32
  {
    int blocks = (MT / 64) * ((3 * D_) / 64);  // 64 * 24
    wmma_gemm_kernel<<<blocks, 128, 0, stream>>>(xb, WqkvT, nullptr, qkvraw,
                                                 MT, 3 * D_, D_);
  }
  // 4) RoPE (full 512-dim pairs d,d+256) + head permute + V transpose
  {
    int n = B_ * T_ * (D_ / 2);
    rope_permute_kernel<<<(n + 255) / 256, 256, 0, stream>>>(qkvraw, qbuf, kbuf,
                                                             vTbuf);
  }
  // 5) sliding-window attention (TDM-staged Q/K tiles)
  {
    dim3 grid(T_ / 16, B_ * H_);
    attn_kernel<<<grid, 32, 0, stream>>>(qbuf, kbuf, vTbuf, ao);
  }
  // 6) output projection + bias: (4096x512) @ (512x512) + blin -> f32 out
  {
    int blocks = (MT / 64) * (D_ / 64);  // 64 * 8
    wmma_gemm_kernel<<<blocks, 128, 0, stream>>>(ao, WlinT, blin, out, MT, D_,
                                                 D_);
  }
}
